// BiLSTM_18184891531327
// MI455X (gfx1250) — compile-verified, matrix operand order checked by reading, not run on hardware
//
#include <hip/hip_runtime.h>

// ---------------- constants ----------------
#define S_   256
#define B_   32
#define W_   16
#define EW_  300
#define EC_  32
#define FOUT_ 128
#define H_   256
#define T_   20
#define NH_  8
#define DH_  64
#define K0P_ 448   // padded EW_+FOUT_ (428 -> 448, mult of 32)

typedef __bf16 bf16;
typedef __attribute__((ext_vector_type(16))) __bf16 v16bf;
typedef __attribute__((ext_vector_type(8)))  float  v8f;
typedef __attribute__((ext_vector_type(4)))  unsigned int v4u;

// Load a 16x32 bf16 WMMA fragment (A layout; B^T from row-major [N,K] uses the
// same addressing with the N row). Per ISA 05_wmma: lane l holds row (l&15),
// k-base += 8 for lanes 16..31; elements 0..7 = k..k+7, elements 8..15 = k+16..k+23.
__device__ __forceinline__ v16bf load_frag(const bf16* p) {
  union { v16bf v; v4u q[2]; } u;
  u.q[0] = *(const v4u*)(p);
  u.q[1] = *(const v4u*)(p + 16);
  return u.v;
}

// ---------------- weight convert / pad (f32 -> bf16, zero pad) ----------------
__global__ void pad_bf16_kernel(const float* __restrict__ src, bf16* __restrict__ dst,
                                int R, int Cs, int Rd, int Cd) {
  long long i = (long long)blockIdx.x * blockDim.x + threadIdx.x;
  long long tot = (long long)Rd * Cd;
  if (i >= tot) return;
  int r = (int)(i / Cd), c = (int)(i % Cd);
  float v = (r < R && c < Cs) ? src[(long long)r * Cs + c] : 0.f;
  dst[i] = (bf16)v;
}

__global__ void f32_to_bf16_kernel(const float* __restrict__ s, bf16* __restrict__ d, long long n) {
  long long i = (long long)blockIdx.x * blockDim.x + threadIdx.x;
  if (i < n) d[i] = (bf16)s[i];
}

// ---------------- embedding + char CNN -> X0 bf16 [S*B, 448] ----------------
__global__ void __launch_bounds__(128) embed_char_kernel(
    const int* __restrict__ words, const int* __restrict__ chars,
    const float* __restrict__ wemb, const float* __restrict__ cemb,
    const float* __restrict__ convw, const float* __restrict__ convb,
    bf16* __restrict__ X0) {
  int row = blockIdx.x;                 // row = s*B + b
  int s = row >> 5, b = row & 31;
  int t = threadIdx.x;
  __shared__ float ce[W_][EC_];
  __shared__ int cidx[W_];
  if (t < W_) cidx[t] = chars[((b * S_) + s) * W_ + t];
  __syncthreads();
  for (int i = t; i < W_ * EC_; i += 128) {
    int w = i >> 5, e = i & 31;
    ce[w][e] = cemb[cidx[w] * EC_ + e];
  }
  int wid = words[row];
  bf16* xrow = X0 + (long long)row * K0P_;
  for (int c = t; c < EW_; c += 128)
    xrow[c] = (bf16)wemb[(long long)wid * EW_ + c];
  __syncthreads();
  // grouped conv1d, groups=EC, NF=4 filters/group, K=3, valid, max-pool over width
  int ec = t >> 2;
  float w0 = convw[t * 3 + 0], w1 = convw[t * 3 + 1], w2 = convw[t * 3 + 2];
  float bb = convb[t];
  float m = -3.4e38f;
  #pragma unroll
  for (int p = 0; p < W_ - 2; ++p) {
    float v = bb + w0 * ce[p][ec] + w1 * ce[p + 1][ec] + w2 * ce[p + 2][ec];
    m = fmaxf(m, v);
  }
  xrow[EW_ + t] = (bf16)m;
  for (int c = EW_ + FOUT_ + t; c < K0P_; c += 128) xrow[c] = (bf16)0.f;
}

// ---------------- generic batched GEMM: C = alpha*(A @ W^T) + bias(+bias2) ----
// A: bf16 [M,K] row-major (lda), W: bf16 [N,K] row-major (ldw), C: f32 (ldc).
__global__ void __launch_bounds__(256) gemm_bf16_kernel(
    const bf16* __restrict__ A, const bf16* __restrict__ Wm,
    const float* __restrict__ bias, const float* __restrict__ bias2,
    float* __restrict__ C, int M, int N, int K, int lda, int ldw, int ldc,
    long long sA, long long sW, long long sC, float alpha) {
  long long z = blockIdx.z;
  A += z * sA; Wm += z * sW; C += z * sC;
  int lane = threadIdx.x & 31;
  int wave = threadIdx.x >> 5;
  int n0 = (blockIdx.x * 8 + wave) * 16;
  int m0 = blockIdx.y * 16;
  if (n0 >= N) return;  // wave-uniform exit (EXEC stays all-ones for WMMA)
  const bf16* arow = A + (long long)(m0 + (lane & 15)) * lda + ((lane >> 4) << 3);
  const bf16* wrow = Wm + (long long)(n0 + (lane & 15)) * ldw + ((lane >> 4) << 3);
  v8f acc = {0.f, 0.f, 0.f, 0.f, 0.f, 0.f, 0.f, 0.f};
  for (int k0 = 0; k0 < K; k0 += 32) {
    __builtin_prefetch(wrow + k0 + 128, 0, 1);
    v16bf af = load_frag(arow + k0);
    v16bf bfr = load_frag(wrow + k0);
    acc = __builtin_amdgcn_wmma_f32_16x16x32_bf16(false, af, false, bfr,
                                                  (short)0, acc, false, false);
  }
  int ncol = n0 + (lane & 15);
  if (ncol < N) {
    float badd = (bias ? bias[ncol] : 0.f) + (bias2 ? bias2[ncol] : 0.f);
    int rbase = m0 + ((lane >> 4) << 3);   // C layout: vgpr r -> row r (+8 upper half)
    #pragma unroll
    for (int r = 0; r < 8; ++r)
      C[(long long)(rbase + r) * ldc + ncol] = acc[r] * alpha + badd;
  }
}

// ---------------- one BiLSTM direction, whole sequence, one workgroup --------
// Xg: precomputed x@W_ih^T + b_ih + b_hh, f32 [S][B][4H]. Whh: bf16 [4H][H].
// hcat: bf16 [S][B][2H], this direction writes columns [dir*H, dir*H+H).
#define HP_ 264   // padded LDS row stride for h (multiple of 8 halves)
__global__ void __launch_bounds__(1024) lstm_dir_kernel(
    const float* __restrict__ Xg0, const float* __restrict__ Xg1,
    const bf16* __restrict__ Whh0, const bf16* __restrict__ Whh1,
    bf16* __restrict__ hcat) {
  int dir = blockIdx.x;
  const float* Xg = dir ? Xg1 : Xg0;
  const bf16* Whh = dir ? Whh1 : Whh0;
  extern __shared__ char smem[];
  bf16*  hprev = (bf16*)smem;                                   // [B][HP_]
  float* cst   = (float*)(smem + B_ * HP_ * 2);                 // [B][H]
  float* gbuf  = (float*)(smem + B_ * HP_ * 2 + B_ * H_ * 4);   // [B][4H]
  int tid = threadIdx.x, lane = tid & 31, wave = tid >> 5;
  for (int i = tid; i < B_ * HP_; i += 1024) hprev[i] = (bf16)0.f;
  for (int i = tid; i < B_ * H_; i += 1024) cst[i] = 0.f;
  __syncthreads();
  for (int step = 0; step < S_; ++step) {
    int s = dir ? (S_ - 1 - step) : step;
    const float* xg = Xg + (long long)s * (B_ * 4 * H_);
    // gates g[B=32, 4H=1024] = xg + hprev @ Whh^T : 128 WMMA tiles over 32 waves
    #pragma unroll
    for (int it = 0; it < 4; ++it) {
      int tile = wave + 32 * it;
      int m0 = (tile >> 6) << 4;      // 0 or 16
      int n0 = (tile & 63) << 4;
      int ncol = n0 + (lane & 15);
      int rbase = m0 + ((lane >> 4) << 3);
      v8f acc;
      #pragma unroll
      for (int r = 0; r < 8; ++r) acc[r] = xg[(rbase + r) * 1024 + ncol];
      const bf16* arow = hprev + (m0 + (lane & 15)) * HP_ + ((lane >> 4) << 3);
      const bf16* wrow = Whh + (long long)(n0 + (lane & 15)) * H_ + ((lane >> 4) << 3);
      #pragma unroll
      for (int k0 = 0; k0 < H_; k0 += 32) {
        v16bf af = load_frag(arow + k0);
        v16bf bfr = load_frag(wrow + k0);
        acc = __builtin_amdgcn_wmma_f32_16x16x32_bf16(false, af, false, bfr,
                                                      (short)0, acc, false, false);
      }
      #pragma unroll
      for (int r = 0; r < 8; ++r) gbuf[(rbase + r) * 1024 + ncol] = acc[r];
    }
    __syncthreads();
    // elementwise gates (PyTorch order i,f,g,o)
    for (int i = tid; i < B_ * H_; i += 1024) {
      int b = i >> 8, hh = i & 255;
      float gi = gbuf[b * 1024 + hh];
      float gf = gbuf[b * 1024 + 256 + hh];
      float gg = gbuf[b * 1024 + 512 + hh];
      float go = gbuf[b * 1024 + 768 + hh];
      float c = (1.f / (1.f + __expf(-gf))) * cst[i] +
                (1.f / (1.f + __expf(-gi))) * tanhf(gg);
      float h = (1.f / (1.f + __expf(-go))) * tanhf(c);
      cst[i] = c;
      hprev[b * HP_ + hh] = (bf16)h;
      hcat[((long long)s * B_ + b) * (2 * H_) + dir * H_ + hh] = (bf16)h;
    }
    __syncthreads();
  }
}

// ---------------- qkv split: f32 [S,B,3*2H] -> q,k bf16 [B*NH,S,64], vT [B*NH,64,S]
__global__ void qkv_split_kernel(const float* __restrict__ qkv,
                                 bf16* __restrict__ qb, bf16* __restrict__ kb,
                                 bf16* __restrict__ vT) {
  long long i = (long long)blockIdx.x * blockDim.x + threadIdx.x;
  if (i >= (long long)S_ * B_ * NH_ * DH_) return;
  int d = (int)(i & 63);
  int h = (int)((i >> 6) & 7);
  int b = (int)((i >> 9) & 31);
  int s = (int)(i >> 14);
  const float* base = qkv + ((long long)s * B_ + b) * (3 * 2 * H_);
  int z = b * NH_ + h;
  qb[((long long)z * S_ + s) * DH_ + d] = (bf16)base[h * DH_ + d];
  kb[((long long)z * S_ + s) * DH_ + d] = (bf16)base[512 + h * DH_ + d];
  vT[((long long)z * DH_ + d) * S_ + s] = (bf16)base[1024 + h * DH_ + d];
}

// ---------------- softmax over last dim of scores [B*NH, S, S] (in place) ----
__global__ void __launch_bounds__(256) softmax_kernel(float* __restrict__ sc,
                                                      bf16* __restrict__ pb) {
  long long row = blockIdx.x;
  float* p = sc + row * S_;
  int t = threadIdx.x;
  __shared__ float red[256];
  float v = p[t];
  red[t] = v; __syncthreads();
  for (int o = 128; o > 0; o >>= 1) { if (t < o) red[t] = fmaxf(red[t], red[t + o]); __syncthreads(); }
  float mx = red[0]; __syncthreads();
  float e = __expf(v - mx);
  red[t] = e; __syncthreads();
  for (int o = 128; o > 0; o >>= 1) { if (t < o) red[t] += red[t + o]; __syncthreads(); }
  float prob = e / red[0];
  p[t] = prob;
  pb[row * S_ + t] = (bf16)prob;
}

// ---------------- attn_weight = mean over heads -> d_out region --------------
__global__ void headmean_kernel(const float* __restrict__ sc, float* __restrict__ out) {
  long long i = (long long)blockIdx.x * blockDim.x + threadIdx.x;
  if (i >= (long long)B_ * S_ * S_) return;
  int k = (int)(i & 255);
  int q = (int)((i >> 8) & 255);
  int b = (int)(i >> 16);
  float acc = 0.f;
  #pragma unroll
  for (int h = 0; h < NH_; ++h)
    acc += sc[(((long long)(b * NH_ + h) * S_) + q) * S_ + k];
  out[i] = acc * (1.f / NH_);
}

// ---------------- ctx [B*NH,S,64] f32 -> attn_in bf16 [S,B,512] --------------
__global__ void ctx_pack_kernel(const float* __restrict__ ctx, bf16* __restrict__ ai) {
  long long i = (long long)blockIdx.x * blockDim.x + threadIdx.x;
  if (i >= (long long)S_ * B_ * NH_ * DH_) return;
  int d = (int)(i & 63);
  int h = (int)((i >> 6) & 7);
  int b = (int)((i >> 9) & 31);
  int s = (int)(i >> 14);
  ai[((long long)s * B_ + b) * (2 * H_) + h * DH_ + d] =
      (bf16)ctx[(((long long)(b * NH_ + h) * S_) + s) * DH_ + d];
}

// ---------------- CRF: Viterbi decode + NLL, one wave, lane = batch ----------
__global__ void __launch_bounds__(32) crf_kernel(
    const float* __restrict__ emis, const int* __restrict__ tags,
    const float* __restrict__ startt, const float* __restrict__ endt,
    const float* __restrict__ trans, int* __restrict__ hist,
    float* __restrict__ out) {
  int b = threadIdx.x;
  __shared__ float tr[T_][T_];
  __shared__ float st[T_], en[T_];
  __shared__ float lred[32];
  if (b < T_) { st[b] = startt[b]; en[b] = endt[b]; }
  for (int i = b; i < T_ * T_; i += 32) tr[i / T_][i % T_] = trans[i];
  __syncthreads();
  float vs[T_], fs[T_];
  const float* e0 = emis + b * T_;
  for (int j = 0; j < T_; ++j) { float v = st[j] + e0[j]; vs[j] = v; fs[j] = v; }
  int ptag = tags[b];
  float num = st[ptag] + e0[ptag];
  for (int s = 1; s < S_; ++s) {
    const float* es = emis + ((long long)s * B_ + b) * T_;
    float nv[T_], nf[T_];
    for (int j = 0; j < T_; ++j) {
      float bm = vs[0] + tr[0][j]; int bi = 0;
      float fm = fs[0] + tr[0][j];
      for (int i2 = 1; i2 < T_; ++i2) {
        float cand = vs[i2] + tr[i2][j];
        if (cand > bm) { bm = cand; bi = i2; }
        fm = fmaxf(fm, fs[i2] + tr[i2][j]);
      }
      float se = 0.f;
      for (int i2 = 0; i2 < T_; ++i2) se += __expf(fs[i2] + tr[i2][j] - fm);
      nv[j] = bm + es[j];
      nf[j] = fm + __logf(se) + es[j];
      hist[((long long)s * B_ + b) * T_ + j] = bi;
    }
    for (int j = 0; j < T_; ++j) { vs[j] = nv[j]; fs[j] = nf[j]; }
    int tg = tags[s * B_ + b];
    num += tr[ptag][tg] + es[tg];
    ptag = tg;
  }
  num += en[ptag];
  float bm = vs[0] + en[0]; int last = 0;
  for (int j = 1; j < T_; ++j) { float c = vs[j] + en[j]; if (c > bm) { bm = c; last = j; } }
  float fm = fs[0] + en[0];
  for (int j = 1; j < T_; ++j) fm = fmaxf(fm, fs[j] + en[j]);
  float se = 0.f;
  for (int j = 0; j < T_; ++j) se += __expf(fs[j] + en[j] - fm);
  float den = fm + __logf(se);
  // backtrack -> decode [B,S]
  out[b * S_ + (S_ - 1)] = (float)last;
  int tg = last;
  for (int s = S_ - 1; s >= 1; --s) {
    tg = hist[((long long)s * B_ + b) * T_ + tg];
    out[b * S_ + (s - 1)] = (float)tg;
  }
  lred[b] = num - den;
  __syncthreads();
  if (b == 0) {
    float tot = 0.f;
    for (int i = 0; i < 32; ++i) tot += lred[i];
    out[B_ * S_] = -tot;   // crf_loss at index 8192
  }
}

// ============================ host side ============================
static inline void launch_pad(hipStream_t st, const float* src, bf16* dst,
                              int R, int Cs, int Rd, int Cd) {
  long long tot = (long long)Rd * Cd;
  pad_bf16_kernel<<<(int)((tot + 255) / 256), 256, 0, st>>>(src, dst, R, Cs, Rd, Cd);
}

static inline void launch_gemm(hipStream_t st, const bf16* A, const bf16* Wt,
                               const float* b1, const float* b2, float* C,
                               int M, int N, int K, int lda, int ldw, int ldc,
                               long long sA, long long sW, long long sC,
                               int batch, float alpha) {
  dim3 grid((N + 16 * 8 - 1) / (16 * 8), M / 16, batch);
  gemm_bf16_kernel<<<grid, 256, 0, st>>>(A, Wt, b1, b2, C, M, N, K, lda, ldw, ldc,
                                         sA, sW, sC, alpha);
}

extern "C" void kernel_launch(void* const* d_in, const int* in_sizes, int n_in,
                              void* d_out, int out_size, void* d_ws, size_t ws_size,
                              hipStream_t stream) {
  const int*   words   = (const int*)d_in[0];
  const int*   chars   = (const int*)d_in[1];
  const int*   tags    = (const int*)d_in[2];
  const float* wemb    = (const float*)d_in[3];
  const float* cemb    = (const float*)d_in[4];
  const float* convw   = (const float*)d_in[5];
  const float* convb   = (const float*)d_in[6];
  const float* w_ih[4] = {(const float*)d_in[7],  (const float*)d_in[11],
                          (const float*)d_in[15], (const float*)d_in[19]};
  const float* w_hh[4] = {(const float*)d_in[8],  (const float*)d_in[12],
                          (const float*)d_in[16], (const float*)d_in[20]};
  const float* b_ih[4] = {(const float*)d_in[9],  (const float*)d_in[13],
                          (const float*)d_in[17], (const float*)d_in[21]};
  const float* b_hh[4] = {(const float*)d_in[10], (const float*)d_in[14],
                          (const float*)d_in[18], (const float*)d_in[22]};
  const float* in_proj_w  = (const float*)d_in[23];
  const float* in_proj_b  = (const float*)d_in[24];
  const float* out_proj_w = (const float*)d_in[25];
  const float* out_proj_b = (const float*)d_in[26];
  const float* fc_w       = (const float*)d_in[27];
  const float* fc_b       = (const float*)d_in[28];
  const float* start_tr   = (const float*)d_in[29];
  const float* end_tr     = (const float*)d_in[30];
  const float* trans      = (const float*)d_in[31];
  float* out = (float*)d_out;

  // ---- workspace bump allocator ----
  char* base = (char*)d_ws;
  size_t off = 0;
  auto alloc = [&](size_t bytes) -> char* {
    char* p = base + off;
    off = (off + bytes + 255) & ~(size_t)255;
    return p;
  };
  const int M = S_ * B_;                       // 8192 rows
  bf16* wb_ih0f = (bf16*)alloc((size_t)1024 * K0P_ * 2);
  bf16* wb_ih0b = (bf16*)alloc((size_t)1024 * K0P_ * 2);
  bf16* wb_hh0f = (bf16*)alloc((size_t)1024 * H_ * 2);
  bf16* wb_hh0b = (bf16*)alloc((size_t)1024 * H_ * 2);
  bf16* wb_ih1f = (bf16*)alloc((size_t)1024 * 512 * 2);
  bf16* wb_ih1b = (bf16*)alloc((size_t)1024 * 512 * 2);
  bf16* wb_hh1f = (bf16*)alloc((size_t)1024 * H_ * 2);
  bf16* wb_hh1b = (bf16*)alloc((size_t)1024 * H_ * 2);
  bf16* wb_inp  = (bf16*)alloc((size_t)1536 * 512 * 2);
  bf16* wb_outp = (bf16*)alloc((size_t)512 * 512 * 2);
  bf16* wb_fc   = (bf16*)alloc((size_t)32 * 512 * 2);      // rows padded 20->32
  bf16* X0      = (bf16*)alloc((size_t)M * K0P_ * 2);
  float* Xgf    = (float*)alloc((size_t)M * 1024 * 4);
  float* Xgb    = (float*)alloc((size_t)M * 1024 * 4);
  bf16* hcat0   = (bf16*)alloc((size_t)M * 512 * 2);
  bf16* hcat1   = (bf16*)alloc((size_t)M * 512 * 2);
  float* qkv    = (float*)alloc((size_t)M * 1536 * 4);
  bf16* qb      = (bf16*)alloc((size_t)B_ * NH_ * S_ * DH_ * 2);
  bf16* kb      = (bf16*)alloc((size_t)B_ * NH_ * S_ * DH_ * 2);
  bf16* vT      = (bf16*)alloc((size_t)B_ * NH_ * DH_ * S_ * 2);
  float* scores = (float*)alloc((size_t)B_ * NH_ * S_ * S_ * 4);
  bf16* probs   = (bf16*)alloc((size_t)B_ * NH_ * S_ * S_ * 2);
  float* ctx    = (float*)alloc((size_t)B_ * NH_ * S_ * DH_ * 4);
  bf16* attn_in = (bf16*)alloc((size_t)M * 512 * 2);
  float* ao_f   = (float*)alloc((size_t)M * 512 * 4);
  bf16* ao_b    = (bf16*)alloc((size_t)M * 512 * 2);
  float* emis   = (float*)alloc((size_t)M * T_ * 4);
  int*  hist    = (int*)alloc((size_t)S_ * B_ * T_ * 4);
  (void)in_sizes; (void)n_in; (void)out_size; (void)ws_size;

  // ---- 1. convert / pad weights to bf16 ----
  launch_pad(stream, w_ih[0], wb_ih0f, 1024, EW_ + FOUT_, 1024, K0P_);
  launch_pad(stream, w_ih[1], wb_ih0b, 1024, EW_ + FOUT_, 1024, K0P_);
  launch_pad(stream, w_hh[0], wb_hh0f, 1024, H_, 1024, H_);
  launch_pad(stream, w_hh[1], wb_hh0b, 1024, H_, 1024, H_);
  launch_pad(stream, w_ih[2], wb_ih1f, 1024, 512, 1024, 512);
  launch_pad(stream, w_ih[3], wb_ih1b, 1024, 512, 1024, 512);
  launch_pad(stream, w_hh[2], wb_hh1f, 1024, H_, 1024, H_);
  launch_pad(stream, w_hh[3], wb_hh1b, 1024, H_, 1024, H_);
  launch_pad(stream, in_proj_w, wb_inp, 1536, 512, 1536, 512);
  launch_pad(stream, out_proj_w, wb_outp, 512, 512, 512, 512);
  launch_pad(stream, fc_w, wb_fc, T_, 512, 32, 512);

  // ---- 2. embeddings + char CNN ----
  embed_char_kernel<<<M, 128, 0, stream>>>(words, chars, wemb, cemb, convw, convb, X0);

  // ---- 3. layer 0: input projections + recurrence ----
  launch_gemm(stream, X0, wb_ih0f, b_ih[0], b_hh[0], Xgf,
              M, 1024, K0P_, K0P_, K0P_, 1024, 0, 0, 0, 1, 1.f);
  launch_gemm(stream, X0, wb_ih0b, b_ih[1], b_hh[1], Xgb,
              M, 1024, K0P_, K0P_, K0P_, 1024, 0, 0, 0, 1, 1.f);
  size_t lstm_smem = (size_t)B_ * HP_ * 2 + (size_t)B_ * H_ * 4 + (size_t)B_ * 4 * H_ * 4;
  lstm_dir_kernel<<<2, 1024, lstm_smem, stream>>>(Xgf, Xgb, wb_hh0f, wb_hh0b, hcat0);

  // ---- 4. layer 1 ----
  launch_gemm(stream, hcat0, wb_ih1f, b_ih[2], b_hh[2], Xgf,
              M, 1024, 512, 512, 512, 1024, 0, 0, 0, 1, 1.f);
  launch_gemm(stream, hcat0, wb_ih1b, b_ih[3], b_hh[3], Xgb,
              M, 1024, 512, 512, 512, 1024, 0, 0, 0, 1, 1.f);
  lstm_dir_kernel<<<2, 1024, lstm_smem, stream>>>(Xgf, Xgb, wb_hh1f, wb_hh1b, hcat1);

  // ---- 5. attention ----
  launch_gemm(stream, hcat1, wb_inp, in_proj_b, nullptr, qkv,
              M, 1536, 512, 512, 512, 1536, 0, 0, 0, 1, 1.f);
  {
    long long n = (long long)M * 512;
    qkv_split_kernel<<<(int)((n + 255) / 256), 256, 0, stream>>>(qkv, qb, kb, vT);
  }
  // scores[z] = (Q[z] @ K[z]^T) / sqrt(64), z = b*NH+h
  launch_gemm(stream, qb, kb, nullptr, nullptr, scores,
              S_, S_, DH_, DH_, DH_, S_,
              (long long)S_ * DH_, (long long)S_ * DH_, (long long)S_ * S_,
              B_ * NH_, 0.125f);
  softmax_kernel<<<B_ * NH_ * S_, 256, 0, stream>>>(scores, probs);
  {
    long long n = (long long)B_ * S_ * S_;
    headmean_kernel<<<(int)((n + 255) / 256), 256, 0, stream>>>(scores, out + B_ * S_ + 1);
  }
  // ctx[z] = P[z] @ V[z]  (vT stored [z,64,S] row-major => W[N=64,K=S])
  launch_gemm(stream, probs, vT, nullptr, nullptr, ctx,
              S_, DH_, S_, S_, S_, DH_,
              (long long)S_ * S_, (long long)DH_ * S_, (long long)S_ * DH_,
              B_ * NH_, 1.f);
  {
    long long n = (long long)M * 512;
    ctx_pack_kernel<<<(int)((n + 255) / 256), 256, 0, stream>>>(ctx, attn_in);
  }
  launch_gemm(stream, attn_in, wb_outp, out_proj_b, nullptr, ao_f,
              M, 512, 512, 512, 512, 512, 0, 0, 0, 1, 1.f);
  {
    long long n = (long long)M * 512;
    f32_to_bf16_kernel<<<(int)((n + 255) / 256), 256, 0, stream>>>(ao_f, ao_b, n);
  }
  // ---- 6. emissions ----
  launch_gemm(stream, ao_b, wb_fc, fc_b, nullptr, emis,
              M, T_, 512, 512, 512, T_, 0, 0, 0, 1, 1.f);

  // ---- 7. CRF decode + loss ----
  crf_kernel<<<1, 32, 0, stream>>>(emis, tags, start_tr, end_tr, trans, hist, out);
}
